// VoxelNetLiDAREncoder_43473658970517
// MI455X (gfx1250) — matrix operand order repaired and checked by reading, not compiled, not run on hardware
//
#include <hip/hip_runtime.h>
#include <hip/hip_bf16.h>

typedef __attribute__((ext_vector_type(8)))  float        v8f;
typedef __attribute__((ext_vector_type(8)))  __bf16       v8bf;
typedef __attribute__((ext_vector_type(16))) __bf16       v16bf;
typedef __attribute__((ext_vector_type(4)))  unsigned int v4u;
typedef __attribute__((ext_vector_type(8)))  int          v8i;
typedef __attribute__((ext_vector_type(4)))  int          v4i;

static __device__ __forceinline__ unsigned short f2bf(float f) {
    unsigned u = __float_as_uint(f);
    unsigned r = (u + 0x7FFFu + ((u >> 16) & 1u)) >> 16;   // round-to-nearest-even
    return (unsigned short)r;
}

static __device__ __forceinline__ v8f wmma_bf16(v16bf a, v16bf b, v8f c) {
    return __builtin_amdgcn_wmma_f32_16x16x32_bf16(false, a, false, b, (short)0, c, false, false);
}

// A fragment: 16 contiguous bf16 at p (K offset 16*hi)
static __device__ __forceinline__ v16bf ldfragA(const unsigned short* p) {
    v8bf x = *(const v8bf*)p;
    v8bf y = *(const v8bf*)(p + 8);
    return __builtin_shufflevector(x, y, 0, 1, 2, 3, 4, 5, 6, 7, 8, 9, 10, 11, 12, 13, 14, 15);
}
// B fragment: 16 contiguous bf16 at p (K offset 16*hi)
static __device__ __forceinline__ v16bf ldfragB(const unsigned short* p) {
    v8bf x = *(const v8bf*)p;
    v8bf y = *(const v8bf*)(p + 8);
    return __builtin_shufflevector(x, y, 0, 1, 2, 3, 4, 5, 6, 7, 8, 9, 10, 11, 12, 13, 14, 15);
}

// ---------------------------------------------------------------------------
// Tensor Data Mover: 2-D tile (tile_d1 rows x tile_d0 elems, bf16) -> LDS
// D# packing per CDNA5 ISA ch.8 (group0 128b, group1 256b, groups 2/3 zero)
// ---------------------------------------------------------------------------
static __device__ __forceinline__ void tdm_load_tile_2d(unsigned lds_addr,
                                                        unsigned long long gaddr,
                                                        unsigned tile_d0, unsigned tile_d1,
                                                        unsigned long long stride0_elems) {
#if __has_builtin(__builtin_amdgcn_tensor_load_to_lds)
    v4u g0;
    g0[0] = 1u;                                            // count=1, user mode
    g0[1] = lds_addr;                                      // LDS byte address
    g0[2] = (unsigned)(gaddr & 0xFFFFFFFFull);             // global_addr[31:0]
    g0[3] = (unsigned)((gaddr >> 32) & 0x01FFFFFFull) | 0x80000000u; // addr[56:32] | type=2
    v8i g1;
    g1[0] = (int)(1u << 16);                               // data_size=1 (2 bytes)
    g1[1] = (int)((tile_d0 & 0xFFFFu) << 16);              // tensor_dim0[15:0]
    g1[2] = (int)(((tile_d0 >> 16) & 0xFFFFu) | ((tile_d1 & 0xFFFFu) << 16)); // dim0 hi | dim1 lo
    g1[3] = (int)(((tile_d1 >> 16) & 0xFFFFu) | ((tile_d0 & 0xFFFFu) << 16)); // dim1 hi | tile_dim0
    g1[4] = (int)(tile_d1 & 0xFFFFu);                      // tile_dim1 | tile_dim2=0
    g1[5] = (int)(stride0_elems & 0xFFFFFFFFull);          // tensor_dim0_stride lo
    g1[6] = (int)((stride0_elems >> 32) & 0xFFFFull);      // stride hi | dim1_stride lo=0
    g1[7] = 0;
    v4i z = {0, 0, 0, 0};
#if __clang_major__ >= 23
    v8i z8 = {0, 0, 0, 0, 0, 0, 0, 0};
    __builtin_amdgcn_tensor_load_to_lds(g0, g1, z, z, z8, 0);
#else
    __builtin_amdgcn_tensor_load_to_lds(g0, g1, z, z, 0);
#endif
#endif
}

// ---------------------------------------------------------------------------
// VFE1 stage A: accumulate sum(x)[5] and lower-tri sum(x x^T)[15] over rows
// ---------------------------------------------------------------------------
__global__ __launch_bounds__(256) void vfe1_moments(const float* __restrict__ x,
                                                    int rows, float* __restrict__ acc) {
    float sx[5] = {0.f, 0.f, 0.f, 0.f, 0.f};
    float sxx[15];
#pragma unroll
    for (int i = 0; i < 15; ++i) sxx[i] = 0.f;
    int stride = gridDim.x * blockDim.x;
    for (int r = blockIdx.x * blockDim.x + threadIdx.x; r < rows; r += stride) {
        float v[5];
#pragma unroll
        for (int c = 0; c < 5; ++c) v[c] = x[(size_t)r * 5 + c];
#pragma unroll
        for (int c = 0; c < 5; ++c) sx[c] += v[c];
        int k = 0;
#pragma unroll
        for (int c = 0; c < 5; ++c)
#pragma unroll
            for (int d = 0; d <= c; ++d) sxx[k++] += v[c] * v[d];
    }
    __shared__ float buf[256];
    for (int q = 0; q < 20; ++q) {
        buf[threadIdx.x] = (q < 5) ? sx[q] : sxx[q - 5];
        __syncthreads();
        for (int s = 128; s > 0; s >>= 1) {
            if (threadIdx.x < (unsigned)s) buf[threadIdx.x] += buf[threadIdx.x + s];
            __syncthreads();
        }
        if (threadIdx.x == 0) atomicAdd(acc + q, buf[0]);
        __syncthreads();
    }
}

// VFE1 stage B: per-output-channel BN scale/shift from analytic moments
__global__ void vfe1_params(const float* __restrict__ acc, const float* __restrict__ w1,
                            const float* __restrict__ g1, const float* __restrict__ be1,
                            float* __restrict__ a1, float* __restrict__ c1, int rows) {
    int o = threadIdx.x;
    if (o >= 64) return;
    float inv = 1.0f / (float)rows;
    float xb[5];
#pragma unroll
    for (int c = 0; c < 5; ++c) xb[c] = acc[c] * inv;
    float M[5][5];
    int k = 0;
#pragma unroll
    for (int c = 0; c < 5; ++c)
#pragma unroll
        for (int d = 0; d <= c; ++d) { float m = acc[5 + k] * inv; M[c][d] = m; M[d][c] = m; ++k; }
    float wv[5];
#pragma unroll
    for (int c = 0; c < 5; ++c) wv[c] = w1[o * 5 + c];
    float mu = 0.f;
#pragma unroll
    for (int c = 0; c < 5; ++c) mu += wv[c] * xb[c];
    float ey2 = 0.f;
#pragma unroll
    for (int c = 0; c < 5; ++c)
#pragma unroll
        for (int d = 0; d < 5; ++d) ey2 += wv[c] * wv[d] * M[c][d];
    float var = ey2 - mu * mu;
    float sc = g1[o] * rsqrtf(var + 1e-5f);
    a1[o] = sc;
    c1[o] = be1[o] - mu * sc;
}

// VFE1 stage C: fused linear -> BN -> ReLU -> max over 32 points, emit bf16
__global__ __launch_bounds__(256) void vfe1_forward(const float* __restrict__ x,
                                                    const float* __restrict__ w1,
                                                    const float* __restrict__ a1,
                                                    const float* __restrict__ c1,
                                                    unsigned short* __restrict__ f1bf) {
    __shared__ float xs[4][32][5];
    int flat = threadIdx.y * 64 + threadIdx.x;
    const float* src = x + (size_t)blockIdx.x * 4 * 32 * 5;
    for (int i = flat; i < 640; i += 256) ((float*)xs)[i] = src[i];
    __syncthreads();
    int o = threadIdx.x, ty = threadIdx.y;
    float wv[5];
#pragma unroll
    for (int c = 0; c < 5; ++c) wv[c] = w1[o * 5 + c];
    float sc = a1[o], sh = c1[o];
    float m = 0.f;  // ReLU output >= 0
#pragma unroll 4
    for (int p = 0; p < 32; ++p) {
        float d = xs[ty][p][0] * wv[0] + xs[ty][p][1] * wv[1] + xs[ty][p][2] * wv[2] +
                  xs[ty][p][3] * wv[3] + xs[ty][p][4] * wv[4];
        m = fmaxf(m, fmaxf(d * sc + sh, 0.f));
    }
    int voxel = blockIdx.x * 4 + ty;
    f1bf[(size_t)voxel * 64 + o] = f2bf(m);
}

// ---------------------------------------------------------------------------
// bf16 WMMA GEMM, 2x2 register blocking (32x32 per wave):
//   out[M,N] = A[M,K] * W[N,K]^T   (row-major, K%32==0, M%32==0, N%32==0)
// ---------------------------------------------------------------------------
__global__ __launch_bounds__(256) void gemm_wmma(const unsigned short* __restrict__ A_,
                                                 const unsigned short* __restrict__ W_,
                                                 float* __restrict__ out,
                                                 int M, int N, int K) {
    const int wave = (blockIdx.x * blockDim.x + threadIdx.x) >> 5;
    const int lane = threadIdx.x & 31;
    const int Nt = N >> 5;
    const int mt = wave / Nt, nt = wave % Nt;
    if (mt * 32 >= M) return;
    const int lo = lane & 15, hi = lane >> 4;
    const int r0 = mt * 32 + lo, r1 = r0 + 16;
    const int n0 = nt * 32 + lo, n1 = n0 + 16;
    v8f acc00 = {}, acc01 = {}, acc10 = {}, acc11 = {};
    for (int k0 = 0; k0 < K; k0 += 32) {
        v16bf A0 = ldfragA(A_ + (size_t)r0 * K + k0 + 16 * hi);
        v16bf A1 = ldfragA(A_ + (size_t)r1 * K + k0 + 16 * hi);
        v16bf B0 = ldfragB(W_ + (size_t)n0 * K + k0 + 16 * hi);
        v16bf B1 = ldfragB(W_ + (size_t)n1 * K + k0 + 16 * hi);
        acc00 = wmma_bf16(A0, B0, acc00);
        acc01 = wmma_bf16(A0, B1, acc01);
        acc10 = wmma_bf16(A1, B0, acc10);
        acc11 = wmma_bf16(A1, B1, acc11);
    }
#pragma unroll
    for (int r = 0; r < 8; ++r) {
        int m0 = mt * 32 + r + 8 * hi, m1 = m0 + 16;
        __builtin_nontemporal_store(acc00[r], &out[(size_t)m0 * N + n0]);
        __builtin_nontemporal_store(acc01[r], &out[(size_t)m0 * N + n1]);
        __builtin_nontemporal_store(acc10[r], &out[(size_t)m1 * N + n0]);
        __builtin_nontemporal_store(acc11[r], &out[(size_t)m1 * N + n1]);
    }
}

// ---------------------------------------------------------------------------
// Implicit-GEMM 3x3x3 stride-2 pad-1 conv, 2x2 register blocking.
//   in  : (B, Di, Hi, Wi, Ci) bf16 channels-last
//   wt  : (Co, 27, Ci) bf16 -> double-buffered through LDS via Tensor Data Mover
//   out : (B*Do*Ho*Wo, Co) f32 channels-last (non-temporal stores)
// Each wave owns a 32x32 output tile and two 8KB LDS weight slices.
// Dynamic LDS = 8 waves * 2 * 32oc * 128ic * 2B = 128KB (<= 320KB/WG on CDNA5).
// ---------------------------------------------------------------------------
__global__ __launch_bounds__(256) void conv3d_wmma(const unsigned short* __restrict__ in,
                                                   const unsigned short* __restrict__ wt,
                                                   float* __restrict__ out,
                                                   int B, int Di, int Hi, int Wi, int Ci,
                                                   int Do, int Ho, int Wo, int Co) {
    extern __shared__ unsigned short wlds[];
    const int waveIB = threadIdx.x >> 5;
    const int wave = (blockIdx.x * blockDim.x + threadIdx.x) >> 5;
    const int lane = threadIdx.x & 31;
    const int Nt = Co >> 5;
    const int mt = wave / Nt, nt = wave % Nt;
    const int M = B * Do * Ho * Wo;
    if (mt * 32 >= M) return;
    const int lo = lane & 15, hi = lane >> 4;

    unsigned short* slice = wlds + (size_t)waveIB * (2 * 32 * 128);   // two 4096-elem buffers
    const unsigned ldsbase = (unsigned)(unsigned long long)slice;

    // decode the two output pixels this lane contributes A rows for
    int od_[2], oh_[2], ow_[2], bb_[2];
#pragma unroll
    for (int f = 0; f < 2; ++f) {
        int p = mt * 32 + f * 16 + lo;
        ow_[f] = p % Wo; int t0 = p / Wo;
        oh_[f] = t0 % Ho; t0 /= Ho;
        od_[f] = t0 % Do; bb_[f] = t0 / Do;
    }
    const int oc0 = nt * 32;
    const unsigned long long stride0 = 27ull * (unsigned)Ci;  // elems between oc rows
    const int nch = Ci >> 7;                                  // 128-wide ic chunks per tap
    const int ntiles = 27 * nch;

    auto tile_ga = [&](int tile) -> unsigned long long {
        int tt = tile / nch;
        int cc = (tile - tt * nch) << 7;
        return (unsigned long long)(const void*)(wt + ((size_t)oc0 * 27 + tt) * Ci + cc);
    };

    v8f acc00 = {}, acc01 = {}, acc10 = {}, acc11 = {};
    tdm_load_tile_2d(ldsbase, tile_ga(0), 128u, 32u, stride0);   // prologue: tile 0 -> buf 0
    for (int tile = 0; tile < ntiles; ++tile) {
        const int buf = tile & 1;
        if (tile + 1 < ntiles) {
            tdm_load_tile_2d(ldsbase + (unsigned)((buf ^ 1) * 8192), tile_ga(tile + 1),
                             128u, 32u, stride0);
            __builtin_amdgcn_s_wait_tensorcnt(1);   // oldest (current buf) has landed
        } else {
            __builtin_amdgcn_s_wait_tensorcnt(0);
        }
        const int t = tile / nch;
        const int cc0 = (tile - t * nch) << 7;
        const int kd = t / 9, kh = (t / 3) % 3, kw = t % 3;
        bool inb[2];
        const unsigned short* ab[2];
#pragma unroll
        for (int f = 0; f < 2; ++f) {
            int id = 2 * od_[f] - 1 + kd, ih = 2 * oh_[f] - 1 + kh, iw = 2 * ow_[f] - 1 + kw;
            inb[f] = (id >= 0) & (id < Di) & (ih >= 0) & (ih < Hi) & (iw >= 0) & (iw < Wi);
            ab[f] = in + ((((size_t)bb_[f] * Di + id) * Hi + ih) * Wi + iw) * Ci + cc0;
        }
        unsigned short* mw = slice + buf * 4096;
#pragma unroll
        for (int c0 = 0; c0 < 128; c0 += 32) {
            v16bf A0 = {}, A1 = {};
            if (inb[0]) A0 = ldfragA(ab[0] + c0 + 16 * hi);
            if (inb[1]) A1 = ldfragA(ab[1] + c0 + 16 * hi);
            v16bf B0 = ldfragB(mw + lo * 128 + c0 + 16 * hi);
            v16bf B1 = ldfragB(mw + (16 + lo) * 128 + c0 + 16 * hi);
            acc00 = wmma_bf16(A0, B0, acc00);
            acc01 = wmma_bf16(A0, B1, acc01);
            acc10 = wmma_bf16(A1, B0, acc10);
            acc11 = wmma_bf16(A1, B1, acc11);
        }
    }
    const int n0 = oc0 + lo, n1 = n0 + 16;
#pragma unroll
    for (int r = 0; r < 8; ++r) {
        int m0 = mt * 32 + r + 8 * hi, m1 = m0 + 16;
        __builtin_nontemporal_store(acc00[r], &out[(size_t)m0 * Co + n0]);
        __builtin_nontemporal_store(acc01[r], &out[(size_t)m0 * Co + n1]);
        __builtin_nontemporal_store(acc10[r], &out[(size_t)m1 * Co + n0]);
        __builtin_nontemporal_store(acc11[r], &out[(size_t)m1 * Co + n1]);
    }
}

// ---------------------------------------------------------------------------
// Coalesced per-channel sum/sumsq over channels-last f32 (channel = i % C)
// ---------------------------------------------------------------------------
__global__ __launch_bounds__(256) void chan_accum(const float* __restrict__ y, size_t n, int C,
                                                  float* __restrict__ accum /* 2*C */) {
    const int tid = threadIdx.x;
    const size_t stride = (size_t)gridDim.x * blockDim.x;      // multiple of C (C in {128,256})
    const int c = (int)(((size_t)blockIdx.x * blockDim.x + tid) % C);
    float s = 0.f, s2 = 0.f;
    for (size_t i = (size_t)blockIdx.x * blockDim.x + tid; i < n; i += stride) {
        float v = y[i];
        s += v; s2 += v * v;
    }
    __shared__ float sh0[256], sh1[256];
    sh0[tid] = s; sh1[tid] = s2;
    __syncthreads();
    if (C == 128) {
        if (tid < 128) {
            atomicAdd(accum + 2 * c,     sh0[tid] + sh0[tid + 128]);
            atomicAdd(accum + 2 * c + 1, sh1[tid] + sh1[tid + 128]);
        }
    } else {
        atomicAdd(accum + 2 * c,     s);
        atomicAdd(accum + 2 * c + 1, s2);
    }
}

__global__ void bn_params(const float* __restrict__ accum, float inv_rows, int C,
                          const float* __restrict__ g, const float* __restrict__ be,
                          float* __restrict__ scale, float* __restrict__ shift) {
    int c = blockIdx.x * blockDim.x + threadIdx.x;
    if (c >= C) return;
    float mu = accum[2 * c] * inv_rows;
    float var = accum[2 * c + 1] * inv_rows - mu * mu;
    float sc = g[c] * rsqrtf(var + 1e-5f);
    scale[c] = sc;
    shift[c] = be[c] - mu * sc;
}

// BN + ReLU + f32->bf16, channels-last
__global__ void bn_relu_bf16(const float* __restrict__ y, const float* __restrict__ scale,
                             const float* __restrict__ shift, unsigned short* __restrict__ out,
                             size_t n, int C) {
    size_t stride = (size_t)gridDim.x * blockDim.x;
    for (size_t i = (size_t)blockIdx.x * blockDim.x + threadIdx.x; i < n; i += stride) {
        int c = (int)(i % C);
        float v = fmaf(y[i], scale[c], shift[c]);
        out[i] = f2bf(fmaxf(v, 0.f));
    }
}

// Final BN + ReLU, channels-last f32 -> NCDHW f32 (reference output layout)
__global__ void bn_relu_final(const float* __restrict__ y, const float* __restrict__ scale,
                              const float* __restrict__ shift, float* __restrict__ out,
                              int B, int Do, int Ho, int Wo, int C) {
    size_t n = (size_t)B * Do * Ho * Wo * C;
    size_t stride = (size_t)gridDim.x * blockDim.x;
    for (size_t i = (size_t)blockIdx.x * blockDim.x + threadIdx.x; i < n; i += stride) {
        int c = (int)(i % C);
        size_t sp = i / C;
        int ow = (int)(sp % Wo); size_t t = sp / Wo;
        int oh = (int)(t % Ho); t /= Ho;
        int od = (int)(t % Do); int b = (int)(t / Do);
        float v = fmaxf(fmaf(y[i], scale[c], shift[c]), 0.f);
        out[((((size_t)b * C + c) * Do + od) * Ho + oh) * Wo + ow] = v;
    }
}

// Scatter bf16 voxel features into channels-last dense grid
__global__ void scatter_grid(const unsigned short* __restrict__ f2bf_,
                             const int* __restrict__ coords,
                             unsigned short* __restrict__ grid,
                             int B, int Nv, int C, int D, int H, int W) {
    int v = blockIdx.x;  // 0 .. B*Nv-1
    int d = coords[(size_t)v * 3 + 0];
    int h = coords[(size_t)v * 3 + 1];
    int w = coords[(size_t)v * 3 + 2];
    int b = v / Nv;
    size_t dst = ((((size_t)b * D + d) * H + h) * W + w) * C;
    for (int c = threadIdx.x; c < C; c += blockDim.x)
        grid[dst + c] = f2bf_[(size_t)v * C + c];
}

// f32 -> bf16 elementwise (same layout)
__global__ void mat_to_bf16(const float* __restrict__ w, unsigned short* __restrict__ out, size_t n) {
    size_t stride = (size_t)gridDim.x * blockDim.x;
    for (size_t i = (size_t)blockIdx.x * blockDim.x + threadIdx.x; i < n; i += stride)
        out[i] = f2bf(w[i]);
}

// conv weight (Co, Ci, 27) f32 -> (Co, 27, Ci) bf16
__global__ void conv_w_cvt(const float* __restrict__ w, unsigned short* __restrict__ out,
                           int Co, int Ci) {
    size_t n = (size_t)Co * Ci * 27;
    size_t stride = (size_t)gridDim.x * blockDim.x;
    for (size_t i = (size_t)blockIdx.x * blockDim.x + threadIdx.x; i < n; i += stride) {
        int t = (int)(i % 27); size_t r = i / 27;
        int ic = (int)(r % Ci); int oc = (int)(r / Ci);
        out[((size_t)oc * 27 + t) * Ci + ic] = f2bf(w[i]);
    }
}

// ---------------------------------------------------------------------------
extern "C" void kernel_launch(void* const* d_in, const int* in_sizes, int n_in,
                              void* d_out, int out_size, void* d_ws, size_t ws_size,
                              hipStream_t stream) {
    (void)in_sizes; (void)n_in; (void)out_size; (void)ws_size;
    const float* vf  = (const float*)d_in[0];
    const int*   vc  = (const int*)d_in[1];
    const float* w1  = (const float*)d_in[5];
    const float* g1  = (const float*)d_in[7];
    const float* be1 = (const float*)d_in[8];
    const float* w2  = (const float*)d_in[9];
    const float* g2  = (const float*)d_in[11];
    const float* be2 = (const float*)d_in[12];
    const float* cw1 = (const float*)d_in[13];
    const float* cg1 = (const float*)d_in[15];
    const float* cbe1= (const float*)d_in[16];
    const float* cw2 = (const float*)d_in[17];
    const float* cg2 = (const float*)d_in[19];
    const float* cbe2= (const float*)d_in[20];
    const float* cw3 = (const float*)d_in[21];
    const float* cg3 = (const float*)d_in[23];
    const float* cbe3= (const float*)d_in[24];
    float* out = (float*)d_out;

    const int B = 2, Nv = 12000, C2 = 128;
    const size_t NVOX = (size_t)B * Nv;          // 24000
    const size_t gridElems = (size_t)B * 20 * 128 * 128 * 128;

    char* ws = (char*)d_ws;
    size_t off = 0;
    auto carve = [&](size_t bytes) -> void* {
        void* p = ws + off;
        off = (off + bytes + 255) & ~(size_t)255;
        return p;
    };
    float* acc1 = (float*)carve(20 * 4);
    float* a1   = (float*)carve(64 * 4);
    float* c1   = (float*)carve(64 * 4);
    float* accum = (float*)carve(4 * 512 * 4);   // 4 stages x (2*C<=512) accumulators
    unsigned short* f1bf = (unsigned short*)carve(NVOX * 64 * 2);
    unsigned short* w2bf = (unsigned short*)carve(128 * 64 * 2);
    float* y2   = (float*)carve(NVOX * 128 * 4);
    float* sc2  = (float*)carve(128 * 4);
    float* sh2  = (float*)carve(128 * 4);
    unsigned short* f2bf_ = (unsigned short*)carve(NVOX * 128 * 2);
    unsigned short* grid  = (unsigned short*)carve(gridElems * 2);
    unsigned short* w1c = (unsigned short*)carve((size_t)128 * 27 * 128 * 2);
    unsigned short* w2c = (unsigned short*)carve((size_t)256 * 27 * 128 * 2);
    unsigned short* w3c = (unsigned short*)carve((size_t)256 * 27 * 256 * 2);
    float* ycl = (float*)carve((size_t)81920 * 128 * 4);   // reused for all conv outputs
    unsigned short* act1 = (unsigned short*)carve((size_t)81920 * 128 * 2);
    unsigned short* act2 = (unsigned short*)carve((size_t)10240 * 256 * 2);
    float* scc = (float*)carve(256 * 4);
    float* shc = (float*)carve(256 * 4);

    hipMemsetAsync(acc1, 0, 20 * 4, stream);
    hipMemsetAsync(accum, 0, 4 * 512 * 4, stream);
    hipMemsetAsync(grid, 0, gridElems * 2, stream);

    const unsigned ldsBytes = 8u * 2u * 32u * 128u * 2u;   // 128KB dynamic LDS for conv

    // ---- VFE1 ----
    const int rows1 = B * Nv * 32;               // 768000
    vfe1_moments<<<384, 256, 0, stream>>>(vf, rows1, acc1);
    vfe1_params<<<1, 64, 0, stream>>>(acc1, w1, g1, be1, a1, c1, rows1);
    vfe1_forward<<<(int)(NVOX / 4), dim3(64, 4), 0, stream>>>(vf, w1, a1, c1, f1bf);

    // ---- VFE2: GEMM (24000x64)x(64x128) via WMMA ----
    mat_to_bf16<<<32, 256, 0, stream>>>(w2, w2bf, 128 * 64);
    {
        int waves = (int)(NVOX / 32) * (128 / 32);   // 3000
        gemm_wmma<<<(waves * 32 + 255) / 256, 256, 0, stream>>>(f1bf, w2bf, y2,
                                                                (int)NVOX, 128, 64);
    }
    chan_accum<<<256, 256, 0, stream>>>(y2, NVOX * 128, 128, accum + 0 * 512);
    bn_params<<<1, 128, 0, stream>>>(accum + 0 * 512, 1.f / (float)NVOX, 128, g2, be2, sc2, sh2);
    bn_relu_bf16<<<1024, 256, 0, stream>>>(y2, sc2, sh2, f2bf_, NVOX * 128, 128);

    // ---- scatter into dense channels-last grid ----
    scatter_grid<<<(int)NVOX, 128, 0, stream>>>(f2bf_, vc, grid, B, Nv, C2, 20, 128, 128);

    // ---- conv1: 128->128, (2,20,128,128)->(2,10,64,64) ----
    conv_w_cvt<<<1024, 256, 0, stream>>>(cw1, w1c, 128, 128);
    {
        int M = B * 10 * 64 * 64;                    // 81920
        int waves = (M / 32) * (128 / 32);           // 10240
        conv3d_wmma<<<(waves * 32 + 255) / 256, 256, ldsBytes, stream>>>(
            grid, w1c, ycl, B, 20, 128, 128, 128, 10, 64, 64, 128);
        chan_accum<<<512, 256, 0, stream>>>(ycl, (size_t)M * 128, 128, accum + 1 * 512);
        bn_params<<<1, 128, 0, stream>>>(accum + 1 * 512, 1.f / (float)M, 128, cg1, cbe1, scc, shc);
        bn_relu_bf16<<<2048, 256, 0, stream>>>(ycl, scc, shc, act1, (size_t)M * 128, 128);
    }

    // ---- conv2: 128->256, (2,10,64,64)->(2,5,32,32) ----
    conv_w_cvt<<<1024, 256, 0, stream>>>(cw2, w2c, 256, 128);
    {
        int M = B * 5 * 32 * 32;                     // 10240
        int waves = (M / 32) * (256 / 32);           // 2560
        conv3d_wmma<<<(waves * 32 + 255) / 256, 256, ldsBytes, stream>>>(
            act1, w2c, ycl, B, 10, 64, 64, 128, 5, 32, 32, 256);
        chan_accum<<<256, 256, 0, stream>>>(ycl, (size_t)M * 256, 256, accum + 2 * 512);
        bn_params<<<1, 256, 0, stream>>>(accum + 2 * 512, 1.f / (float)M, 256, cg2, cbe2, scc, shc);
        bn_relu_bf16<<<1024, 256, 0, stream>>>(ycl, scc, shc, act2, (size_t)M * 256, 256);
    }

    // ---- conv3: 256->256, (2,5,32,32)->(2,3,16,16) ----
    conv_w_cvt<<<1024, 256, 0, stream>>>(cw3, w3c, 256, 256);
    {
        int M = B * 3 * 16 * 16;                     // 1536
        int waves = (M / 32) * (256 / 32);           // 384
        conv3d_wmma<<<(waves * 32 + 255) / 256, 256, ldsBytes, stream>>>(
            act2, w3c, ycl, B, 5, 32, 32, 256, 3, 16, 16, 256);
        chan_accum<<<256, 256, 0, stream>>>(ycl, (size_t)M * 256, 256, accum + 3 * 512);
        bn_params<<<1, 256, 0, stream>>>(accum + 3 * 512, 1.f / (float)M, 256, cg3, cbe3, scc, shc);
        bn_relu_final<<<512, 256, 0, stream>>>(ycl, scc, shc, out, B, 3, 16, 16, 256);
    }
}
